// SelfAttention_20633022890127
// MI455X (gfx1250) — compile-verified, matrix operand order checked by reading, not run on hardware
//
#include <hip/hip_runtime.h>

// ---------------------------------------------------------------------------
// Self-attention (SAGAN-style) for MI455X / gfx1250, wave32 + WMMA bf16.
//   B=4, C=512, N=H*W=4096, Cq=64.
// Stage 0: x -> x^T bf16 via LDS 32x32 tile transpose (v_cvt_pk_bf16_f32);
//          [Wq;Wk;Wv] -> bf16.
// Stage 1: fused projection GEMM  -> Kbuf (f^T [B,N,64] bf16),
//                                    Qbuf (g^T [B,N,64] bf16, pre-scaled by
//                                          log2(e) so softmax runs in exp2),
//                                    Vbuf (h   [B,512,N] bf16)
// Stage 2: flash attention, softmax over keys n for each column j,
//          out = gamma * (V beta) + x, f32.
// ---------------------------------------------------------------------------

#define NPIX  4096
#define CIN   512
#define CQ    64
#define BATCH 4
#define COUT  640   // 64 (f) + 64 (g) + 512 (h)
#define LOG2E 1.4426950408889634f

typedef __attribute__((ext_vector_type(16))) __bf16 v16bf;
typedef __attribute__((ext_vector_type(2)))  __bf16 v2bf;
typedef __attribute__((ext_vector_type(8)))  float  v8f;

union Frag16 {                 // one 16x32 (or 32x16) bf16 WMMA operand / lane
    uint4          q[2];
    unsigned int   u[8];
    unsigned short s[16];
    v16bf          v;
};
union Pack8 {
    unsigned short s[8];
    unsigned int   u32[4];
    uint4          u;
};

__device__ __forceinline__ unsigned short f2bf(float f) {
    // round-to-nearest-even f32 -> bf16 (scalar fallback path)
    unsigned int u = __builtin_bit_cast(unsigned int, f);
    u += 0x7FFFu + ((u >> 16) & 1u);
    return (unsigned short)(u >> 16);
}

__device__ __forceinline__ unsigned int pk2bf(float lo, float hi) {
#if __has_builtin(__builtin_amdgcn_cvt_pk_bf16_f32)
    v2bf r = __builtin_amdgcn_cvt_pk_bf16_f32(lo, hi);
    return __builtin_bit_cast(unsigned int, r);
#else
    return (unsigned int)f2bf(lo) | ((unsigned int)f2bf(hi) << 16);
#endif
}

__device__ __forceinline__ float exp2_fast(float x) {
#if __has_builtin(__builtin_amdgcn_exp2f)
    return __builtin_amdgcn_exp2f(x);     // bare v_exp_f32
#else
    return exp2f(x);
#endif
}

// Load a 16-deep-K bf16 fragment for one lane. K-pattern (per ISA 7.12.2):
// lanes 0-15 hold K = {hl*8 .. hl*8+7} u {16+hl*8 ..} relative to kbase.
__device__ __forceinline__ v16bf load_frag(const unsigned short* base, int kbase, int hl) {
    Frag16 f;
    f.q[0] = *reinterpret_cast<const uint4*>(base + kbase + hl * 8);
    f.q[1] = *reinterpret_cast<const uint4*>(base + kbase + 16 + hl * 8);
    return f.v;
}

#define WMMA_BF16(A, B, C) \
    __builtin_amdgcn_wmma_f32_16x16x32_bf16(false, (A), false, (B), (short)0, (C), false, false)

// ---------------------------------------------------------------------------
// Stage 0a: f32 -> bf16 repack (weights), 8 elements / thread, b128 in/out.
// ---------------------------------------------------------------------------
__global__ void __launch_bounds__(256) cvt_kernel(
    const float* __restrict__ src, unsigned short* __restrict__ dst, int n8)
{
    int i = blockIdx.x * blockDim.x + threadIdx.x;
    if (i >= n8) return;
    const float4* s4 = reinterpret_cast<const float4*>(src) + (size_t)i * 2;
    float4 a = s4[0];
    float4 b = s4[1];
    uint4 r;
    r.x = pk2bf(a.x, a.y);
    r.y = pk2bf(a.z, a.w);
    r.z = pk2bf(b.x, b.y);
    r.w = pk2bf(b.z, b.w);
    reinterpret_cast<uint4*>(dst)[i] = r;
}

// ---------------------------------------------------------------------------
// Stage 0b: x [B][C][N] f32 -> x^T [B][N][C] bf16, 32x32 LDS tile transpose.
// Coalesced b128 f32 reads, packed cvt, padded LDS, coalesced 8B stores.
// ---------------------------------------------------------------------------
__global__ void __launch_bounds__(256) cvtT_kernel(
    const float* __restrict__ src, unsigned short* __restrict__ dst)
{
    __shared__ unsigned short tile[32][34];   // +2 pad: odd dword stride

    int blk = blockIdx.x;                     // B * 16 * 128 tiles
    const int b  = blk / (16 * 128);
    int r2 = blk % (16 * 128);
    const int c0 = (r2 / 128) * 32;
    const int n0 = (r2 % 128) * 32;

    const int row  = threadIdx.x >> 3;        // c within tile, 0..31
    const int col4 = (threadIdx.x & 7) * 4;   // n within tile, step 4

    const float4 f = *reinterpret_cast<const float4*>(
        src + ((size_t)b * CIN + c0 + row) * NPIX + n0 + col4);
    *reinterpret_cast<unsigned int*>(&tile[row][col4])     = pk2bf(f.x, f.y);
    *reinterpret_cast<unsigned int*>(&tile[row][col4 + 2]) = pk2bf(f.z, f.w);

    __syncthreads();

    const int nn = threadIdx.x >> 3;          // n within tile, 0..31
    const int cc = (threadIdx.x & 7) * 4;     // c within tile, step 4
    uint2 u;
    u.x = (unsigned int)tile[cc][nn]     | ((unsigned int)tile[cc + 1][nn] << 16);
    u.y = (unsigned int)tile[cc + 2][nn] | ((unsigned int)tile[cc + 3][nn] << 16);
    *reinterpret_cast<uint2*>(
        dst + ((size_t)b * NPIX + n0 + nn) * CIN + c0 + cc) = u;
}

// ---------------------------------------------------------------------------
// Stage 1: projections. Per wave: one 16n x 64o tile of Out[n,o] = X^T W^T.
// o in [0,64)->f (Kbuf), [64,128)->g (Qbuf, *log2e), [128,640)->h (Vbuf, T).
// Inner loop: 2 b128 (A) + 8 b128 (B) loads, 4 WMMAs per K-step of 32.
// ---------------------------------------------------------------------------
__global__ void __launch_bounds__(256) proj_kernel(
    const unsigned short* __restrict__ Xt,    // [B][NPIX][CIN] bf16 (x^T)
    const unsigned short* __restrict__ Wbf,   // [COUT][CIN]    bf16 (Wq;Wk;Wv)
    unsigned short* __restrict__ Kb, unsigned short* __restrict__ Qb,
    unsigned short* __restrict__ Vb)
{
    const int lane = threadIdx.x & 31;
    const int m    = lane & 15;
    const int hl   = lane >> 4;
    const int wid  = blockIdx.x * 8 + (threadIdx.x >> 5);

    const int b   = wid / 2560;          // 2560 waves per batch
    const int rem = wid % 2560;
    const int n0  = (rem / 10) * 16;
    const int og  = rem % 10;
    const int o0  = og * 64;

    // A-operand source: lane's row n = n0 + m, contiguous in c
    const unsigned short* xrow = Xt + ((size_t)b * NPIX + n0 + m) * CIN;

    const unsigned short* wrow[4];
#pragma unroll
    for (int t = 0; t < 4; ++t)
        wrow[t] = Wbf + (size_t)(o0 + t * 16 + m) * CIN;   // lane's column o

    const v8f vzero = {0.f, 0.f, 0.f, 0.f, 0.f, 0.f, 0.f, 0.f};
    v8f acc[4];
#pragma unroll
    for (int t = 0; t < 4; ++t) acc[t] = vzero;

    for (int c0 = 0; c0 < CIN; c0 += 32) {
        v16bf a = load_frag(xrow, c0, hl);          // A = X^T [16n x 32c]
#pragma unroll
        for (int t = 0; t < 4; ++t) {
            v16bf bb = load_frag(wrow[t], c0, hl);  // B = W^T [32c x 16o]
            acc[t] = WMMA_BF16(a, bb, acc[t]);
        }
    }

    // Epilogue. C layout: lane col = o0+t*16+m, rows n = n0 + r + 8*hl.
    if (og < 2) {
        // og==1 is the query side (g): pre-scale by log2(e) so that the
        // attention softmax can use raw v_exp_f32 (base-2) downstream.
        const float sc = (og == 0) ? 1.0f : LOG2E;
#pragma unroll
        for (int t = 0; t < 4; ++t) {
            int o = o0 + t * 16 + m;
            unsigned short* dst = (o < 64)
                ? (Kb + (size_t)b * NPIX * CQ + o)
                : (Qb + (size_t)b * NPIX * CQ + (o - 64));
#pragma unroll
            for (int r = 0; r < 8; ++r) {
                int n = n0 + r + (hl << 3);
                dst[(size_t)n * CQ] = f2bf(acc[t][r] * sc);
            }
        }
    } else {
#pragma unroll
        for (int t = 0; t < 4; ++t) {
            int ov = o0 - 128 + t * 16 + m;
            Pack8 pk;
#pragma unroll
            for (int r = 0; r < 4; ++r)
                pk.u32[r] = pk2bf(acc[t][2 * r], acc[t][2 * r + 1]);
            // 8 consecutive n per lane -> one b128 store into Vbuf[b][ov][n]
            *reinterpret_cast<uint4*>(
                Vb + ((size_t)b * CIN + ov) * NPIX + n0 + (hl << 3)) = pk.u;
        }
    }
}

// ---------------------------------------------------------------------------
// Stage 2: flash attention. Wave = (b, 16-wide j-tile, 128-wide channel slice).
// S^T[n,j] = K x Q^T (WMMA), online softmax (base-2 domain) over n,
// O^T[o,j] += V^T x P. Score C-layout feeds the PV B-operand in-lane.
// ---------------------------------------------------------------------------
__global__ void __launch_bounds__(256) attn_kernel(
    const float* __restrict__ x, const float* __restrict__ gamma,
    const unsigned short* __restrict__ Kb, const unsigned short* __restrict__ Qb,
    const unsigned short* __restrict__ Vb, float* __restrict__ out)
{
    const int lane = threadIdx.x & 31;
    const int m    = lane & 15;
    const int hl   = lane >> 4;
    const int wid  = blockIdx.x * 8 + (threadIdx.x >> 5);

    const int b   = wid >> 10;           // 1024 waves per batch
    const int rem = wid & 1023;
    const int j0  = (rem >> 2) * 16;
    const int o0  = (rem & 3) * 128;

    const float gma = gamma[0];

    // Q (g-projection, pre-scaled by log2e) as B-operand: lane = column j0+m
    const unsigned short* qrow = Qb + ((size_t)b * NPIX + j0 + m) * CQ;
    const v16bf qb0 = load_frag(qrow, 0, hl);
    const v16bf qb1 = load_frag(qrow, 32, hl);

    const v8f vzero = {0.f, 0.f, 0.f, 0.f, 0.f, 0.f, 0.f, 0.f};
    v8f oacc[8];
#pragma unroll
    for (int t = 0; t < 8; ++t) oacc[t] = vzero;

    float mrun = -__builtin_inff();      // running max of base-2 logits
    float lrun = 0.f;                    // running sum of exp2 terms

    for (int nb = 0; nb < NPIX; nb += 64) {
        if (nb + 64 < NPIX)  // pull next key block toward the WGP
            __builtin_prefetch(Kb + ((size_t)b * NPIX + nb + 64 + lane * 2) * CQ, 0, 0);

        // ---- scores: 4 tiles of S^T[n,j], K-dim = q (2 WMMAs each) ----
        v8f s[4];
#pragma unroll
        for (int nt = 0; nt < 4; ++nt) {
            const unsigned short* krow =
                Kb + ((size_t)b * NPIX + nb + nt * 16 + m) * CQ;  // A row = n
            v16bf ka0 = load_frag(krow, 0, hl);
            v16bf ka1 = load_frag(krow, 32, hl);
            v8f z = vzero;
            z = WMMA_BF16(ka0, qb0, z);
            z = WMMA_BF16(ka1, qb1, z);
            s[nt] = z;
        }

        // ---- online softmax over the 64 keys of this block (exp2) ----
        float bm = -__builtin_inff();
#pragma unroll
        for (int nt = 0; nt < 4; ++nt)
#pragma unroll
            for (int r = 0; r < 8; ++r) bm = fmaxf(bm, s[nt][r]);
        bm = fmaxf(bm, __shfl_xor(bm, 16, 32));   // combine lane pair (l, l+16)

        const float mnew = fmaxf(mrun, bm);
        const float corr = exp2_fast(mrun - mnew);

        float lsum = 0.f;
#pragma unroll
        for (int nt = 0; nt < 4; ++nt)
#pragma unroll
            for (int r = 0; r < 8; ++r) {
                float e = exp2_fast(s[nt][r] - mnew);
                s[nt][r] = e;
                lsum += e;
            }
        lsum += __shfl_xor(lsum, 16, 32);
        lrun = lrun * corr + lsum;
        mrun = mnew;

#pragma unroll
        for (int t = 0; t < 8; ++t)
#pragma unroll
            for (int r = 0; r < 8; ++r) oacc[t][r] *= corr;

        // ---- pack P into PV B-operands (pure in-lane packed cvt) ----
        Frag16 pb0, pb1;
#pragma unroll
        for (int v = 0; v < 8; ++v) {
            const int e = (v & 3) * 2;
            const v8f sa = (v < 4) ? s[0] : s[1];
            const v8f sb = (v < 4) ? s[2] : s[3];
            pb0.u[v] = pk2bf(sa[e], sa[e + 1]);
            pb1.u[v] = pk2bf(sb[e], sb[e + 1]);
        }

        // ---- O^T[o,j] += V^T[o,n] x P[n,j] ----
#pragma unroll
        for (int ot = 0; ot < 8; ++ot) {
            const unsigned short* vrow =
                Vb + ((size_t)b * CIN + o0 + ot * 16 + m) * (size_t)NPIX + nb;
            v16bf va0 = load_frag(vrow, 0, hl);
            v16bf va1 = load_frag(vrow, 32, hl);
            oacc[ot] = WMMA_BF16(va0, pb0.v, oacc[ot]);
            oacc[ot] = WMMA_BF16(va1, pb1.v, oacc[ot]);
        }
    }

    // ---- epilogue: out = gamma * O/l + x ----
    const float ginv = gma / lrun;
#pragma unroll
    for (int ot = 0; ot < 8; ++ot)
#pragma unroll
        for (int r = 0; r < 8; ++r) {
            int o = o0 + ot * 16 + r + (hl << 3);
            size_t idx = ((size_t)b * CIN + o) * NPIX + j0 + m;
            out[idx] = fmaf(ginv, oacc[ot][r], x[idx]);
        }
}

// ---------------------------------------------------------------------------
extern "C" void kernel_launch(void* const* d_in, const int* in_sizes, int n_in,
                              void* d_out, int out_size, void* d_ws, size_t ws_size,
                              hipStream_t stream) {
    const float* x     = (const float*)d_in[0];
    const float* Wq    = (const float*)d_in[1];
    const float* Wk    = (const float*)d_in[2];
    const float* Wv    = (const float*)d_in[3];
    const float* gamma = (const float*)d_in[4];
    float* out = (float*)d_out;

    unsigned short* Kb  = (unsigned short*)d_ws;                   //  2 MB
    unsigned short* Qb  = Kb  + (size_t)BATCH * NPIX * CQ;         //  2 MB
    unsigned short* Vb  = Qb  + (size_t)BATCH * NPIX * CQ;         // 16 MB
    unsigned short* Xt  = Vb  + (size_t)BATCH * CIN * NPIX;        // 16 MB
    unsigned short* Wbf = Xt  + (size_t)BATCH * CIN * NPIX;        // 0.64 MB

    // Stage 0: bf16 repack (weights) + bf16 transpose (x)
    {
        cvtT_kernel<<<BATCH * 16 * 128, 256, 0, stream>>>(x, Xt);
        int wq8 = CQ * CIN / 8;                           // 4096
        cvt_kernel<<<wq8 / 256, 256, 0, stream>>>(Wq, Wbf, wq8);
        cvt_kernel<<<wq8 / 256, 256, 0, stream>>>(Wk, Wbf + (size_t)CQ * CIN, wq8);
        int wv8 = CIN * CIN / 8;                          // 32768
        cvt_kernel<<<wv8 / 256, 256, 0, stream>>>(Wv, Wbf + (size_t)2 * CQ * CIN, wv8);
    }

    // Stage 1: 10240 waves (B * 256 n-tiles * 10 o-groups), 8 waves/WG
    proj_kernel<<<1280, 256, 0, stream>>>(Xt, Wbf, Kb, Qb, Vb);
    // Stage 2: 4096 waves (B * 256 j-tiles * 4 channel slices), 8 waves/WG
    attn_kernel<<<512, 256, 0, stream>>>(x, gamma, Kb, Qb, Vb, out);
}
